// Informer_7026566496389
// MI455X (gfx1250) — matrix operand order repaired and checked by reading, not compiled, not run on hardware
//
#include <hip/hip_runtime.h>
#include <hip/hip_bf16.h>
#include <math.h>
#include <stdint.h>

// ---------------- model constants ----------------
constexpr int B    = 16;
constexpr int L    = 512;
constexpr int C_IN = 12;
constexpr int D    = 512;
constexpr int H    = 8;
constexpr int E    = 64;     // D / H
constexpr int DFF  = 2048;
constexpr int NL   = 3;
constexpr int NC   = 10;
constexpr int U    = 35;     // min(5*ceil(ln 512), 512)
constexpr int M    = B * L;  // 8192 rows

// GEMM tiling
constexpr int BM = 128;
constexpr int BN = 64;
constexpr int BK = 64;

// ---------------- vector types ----------------
typedef __attribute__((ext_vector_type(16))) __bf16 v16bf;
typedef __attribute__((ext_vector_type(8)))  float  v8f;
typedef __attribute__((ext_vector_type(4)))  unsigned int u32x4;
typedef __attribute__((ext_vector_type(8)))  int          i32x8;
typedef __attribute__((ext_vector_type(4)))  int          i32x4;

__device__ __forceinline__ unsigned short f2bf(float f) {
    unsigned int x = __float_as_uint(f);
    unsigned int r = x + 0x7FFFu + ((x >> 16) & 1u);   // round-to-nearest-even
    return (unsigned short)(r >> 16);
}
__device__ __forceinline__ float gelu_exact(float x) {
    return 0.5f * x * (1.0f + erff(x * 0.70710678118654752f));
}
__device__ __forceinline__ unsigned lds_off(const void* p) {
    // LDS aperture: flat addr low 32 bits are the LDS byte offset
    return (unsigned)(uintptr_t)p;
}

// ---- Tensor Data Mover: 2D tile (rows x row_len bf16) from global to LDS ----
// D# per CDNA5 ISA ch.8: group0 = {count/flags, lds_addr, global_addr, type=2},
// group1 = {data_size=2B, tensor_dim0/1, tile_dim0/1, tensor_dim0_stride}.
__device__ __forceinline__ void tdm_load_2d(unsigned lds_addr, const void* gptr,
                                            unsigned row_len, unsigned rows,
                                            unsigned tens_d0, unsigned tens_d1,
                                            unsigned long long stride_elems) {
    unsigned long long ga = (unsigned long long)(uintptr_t)gptr;
    u32x4 g0;
    g0[0] = 1u;                                           // count=1, user descriptor
    g0[1] = lds_addr;                                     // bits 63:32 lds_addr
    g0[2] = (unsigned)(ga & 0xffffffffu);                 // global_addr[31:0]
    g0[3] = (unsigned)((ga >> 32) & 0x01ffffffu)          // global_addr[56:32]
          | (2u << 30);                                   // type = 2 ("image")
    i32x8 g1;
    g1[0] = (int)(1u << 16);                              // data_size=1 -> 2 bytes; wg_mask=0
    g1[1] = (int)((tens_d0 & 0xffffu) << 16);             // tensor_dim0[15:0] @ bits63:48
    g1[2] = (int)((tens_d0 >> 16) | ((tens_d1 & 0xffffu) << 16)); // d0[31:16] | d1[15:0]
    g1[3] = (int)((tens_d1 >> 16) | (row_len << 16));     // d1[31:16] | tile_dim0
    g1[4] = (int)(rows & 0xffffu);                        // tile_dim1 (tile_dim2 = 0)
    g1[5] = (int)(unsigned)(stride_elems & 0xffffffffu);  // tensor_dim0_stride[31:0]
    g1[6] = (int)(unsigned)((stride_elems >> 32) & 0xffffu); // stride[47:32]
    g1[7] = 0;
    i32x4 z4 = {0, 0, 0, 0};
#if defined(__clang_major__) && (__clang_major__ >= 23)
    i32x8 z8 = {0, 0, 0, 0, 0, 0, 0, 0};
    __builtin_amdgcn_tensor_load_to_lds(g0, g1, z4, z4, z8, 0);
#else
    __builtin_amdgcn_tensor_load_to_lds(g0, g1, z4, z4, 0);
#endif
}

// ---------------- fp32 -> bf16 convert ----------------
__global__ __launch_bounds__(256) void cvt_bf16_kernel(const float* __restrict__ src,
                                                       unsigned short* __restrict__ dst, int n) {
    int i = blockIdx.x * 256 + threadIdx.x;
    if (i < n) dst[i] = f2bf(src[i]);
}

// ---------------- token conv + positional embedding ----------------
__global__ __launch_bounds__(256) void embed_kernel(const float* __restrict__ x_enc,
                                                    const float* __restrict__ tok_w,
                                                    float* __restrict__ xf,
                                                    unsigned short* __restrict__ xb) {
    int idx = blockIdx.x * 256 + threadIdx.x;          // over M*D
    if (idx >= M * D) return;
    int d   = idx & (D - 1);
    int row = idx >> 9;                                // b*L + l
    int l   = row & (L - 1);
    int b   = row >> 9;

    float acc = 0.f;
    #pragma unroll
    for (int j = 0; j < 3; ++j) {
        int lj = (l + j - 1 + L) & (L - 1);            // circular pad
        const float* xr = x_enc + ((size_t)(b * L + lj)) * C_IN;
        const float* wr = tok_w + ((size_t)d * C_IN) * 3 + j;
        #pragma unroll
        for (int c = 0; c < C_IN; ++c) acc += xr[c] * wr[c * 3];
    }
    int   i2  = d >> 1;
    float div = expf((float)(2 * i2) * (-9.210340371976184f / (float)D)); // -ln(10000)/D
    float pe  = (d & 1) ? cosf((float)l * div) : sinf((float)l * div);

    float v = acc + pe;
    xf[idx] = v;
    xb[idx] = f2bf(v);
}

// ---------------- TDM + LDS staged bf16 WMMA GEMM ----------------
//  C(MxN) = A(MxK) @ Bw(NxK)^T + bias [+gelu]; block tile 128x64, BK=64,
//  double-buffered LDS fed by TENSOR_LOAD_TO_LDS, 4 accumulators / wave.
__global__ __launch_bounds__(256) void gemm_bf16_wmma(const unsigned short* __restrict__ A,
                                                      const unsigned short* __restrict__ Bw,
                                                      const float* __restrict__ bias,
                                                      float* __restrict__ Cf,
                                                      unsigned short* __restrict__ Cb,
                                                      int Mdim, int Ndim, int K,
                                                      int fuse_gelu) {
    __shared__ unsigned short As[2][BM * BK];          // 2 x 16 KB
    __shared__ unsigned short Bs[2][BN * BK];          // 2 x  8 KB

    const int nbn = Ndim / BN;
    const int bm0 = (blockIdx.x / nbn) * BM;
    const int bn0 = (blockIdx.x % nbn) * BN;

    const int wave = threadIdx.x >> 5;
    const int lane = threadIdx.x & 31;
    const int wm   = (wave & 3) * 32;                  // M offset inside block tile
    const int wn   = (wave >> 2) * 32;                 // N offset inside block tile
    const int lo16 = (lane < 16);
    const int lr   = lane & 15;
    const int ka   = lo16 ? 0 : 16;                    // A-fragment K phase (contiguous, matches B)
    const int kb   = lo16 ? 0 : 16;                    // B-fragment K phase (contiguous, matches A)

    v8f acc00 = {}, acc01 = {}, acc10 = {}, acc11 = {};

    const int nch = K / BK;
    if (wave == 0) {                                   // wave 0 drives the TDM
        tdm_load_2d(lds_off(&As[0][0]), A  + (size_t)bm0 * K, BK, BM, K, Mdim, K);
        tdm_load_2d(lds_off(&Bs[0][0]), Bw + (size_t)bn0 * K, BK, BN, K, Ndim, K);
    }

    for (int c = 0; c < nch; ++c) {
        const int cur = c & 1;
        if (wave == 0) {
            if (c + 1 < nch) {                         // prefetch next chunk, then drain cur
                const int nxt = (c + 1) & 1;
                tdm_load_2d(lds_off(&As[nxt][0]),
                            A  + (size_t)bm0 * K + (size_t)(c + 1) * BK, BK, BM, K, Mdim, K);
                tdm_load_2d(lds_off(&Bs[nxt][0]),
                            Bw + (size_t)bn0 * K + (size_t)(c + 1) * BK, BK, BN, K, Ndim, K);
                __builtin_amdgcn_s_wait_tensorcnt(2);
            } else {
                __builtin_amdgcn_s_wait_tensorcnt(0);
            }
        }
        __syncthreads();                               // chunk c resident for all waves

        const unsigned short* Ab = &As[cur][0];
        const unsigned short* Bb = &Bs[cur][0];
        #pragma unroll
        for (int kk = 0; kk < BK; kk += 32) {
            union { unsigned int u[8]; v16bf v; } a0, a1, b0, b1;
            const unsigned short* ap0 = Ab + (wm + lr)      * BK + kk + ka;
            const unsigned short* ap1 = Ab + (wm + 16 + lr) * BK + kk + ka;
            const unsigned short* bp0 = Bb + (wn + lr)      * BK + kk + kb;
            const unsigned short* bp1 = Bb + (wn + 16 + lr) * BK + kk + kb;
            #pragma unroll
            for (int i = 0; i < 4; ++i) {
                a0.u[i]     = *(const unsigned int*)(ap0 + 2 * i);
                a0.u[4 + i] = *(const unsigned int*)(ap0 + 8 + 2 * i);
                a1.u[i]     = *(const unsigned int*)(ap1 + 2 * i);
                a1.u[4 + i] = *(const unsigned int*)(ap1 + 8 + 2 * i);
            }
            #pragma unroll
            for (int i = 0; i < 8; ++i) {
                b0.u[i] = *(const unsigned int*)(bp0 + 2 * i);
                b1.u[i] = *(const unsigned int*)(bp1 + 2 * i);
            }
            acc00 = __builtin_amdgcn_wmma_f32_16x16x32_bf16(false, a0.v, false, b0.v,
                                                            (short)0, acc00, false, false);
            acc01 = __builtin_amdgcn_wmma_f32_16x16x32_bf16(false, a0.v, false, b1.v,
                                                            (short)0, acc01, false, false);
            acc10 = __builtin_amdgcn_wmma_f32_16x16x32_bf16(false, a1.v, false, b0.v,
                                                            (short)0, acc10, false, false);
            acc11 = __builtin_amdgcn_wmma_f32_16x16x32_bf16(false, a1.v, false, b1.v,
                                                            (short)0, acc11, false, false);
        }
        __syncthreads();                               // all waves done before buffer reuse
    }

    // ---- epilogue: bias (+gelu), fp32 and/or bf16 stores, ISA C-layout ----
    auto store_tile = [&](const v8f& acc, int mbase, int nbase) {
        const int   nn  = nbase + lr;
        const float bs  = bias ? bias[nn] : 0.f;
        const int   mr0 = mbase + (lo16 ? 0 : 8);
        #pragma unroll
        for (int rr = 0; rr < 8; ++rr) {
            float v = acc[rr] + bs;
            if (fuse_gelu) v = gelu_exact(v);
            size_t o = (size_t)(mr0 + rr) * Ndim + nn;
            if (Cf) Cf[o] = v;
            if (Cb) Cb[o] = f2bf(v);
        }
    };
    store_tile(acc00, bm0 + wm,      bn0 + wn);
    store_tile(acc01, bm0 + wm,      bn0 + wn + 16);
    store_tile(acc10, bm0 + wm + 16, bn0 + wn);
    store_tile(acc11, bm0 + wm + 16, bn0 + wn + 16);
}

// ---------------- ProbSparse attention ----------------
__global__ __launch_bounds__(256) void attn_metric_kernel(const float* __restrict__ q,
                                                          const float* __restrict__ k,
                                                          const int* __restrict__ samp, // (L,U)
                                                          float* __restrict__ Mval) {
    int tid = blockIdx.x * 256 + threadIdx.x;          // B*H*L
    if (tid >= B * H * L) return;
    int l = tid & (L - 1);
    int h = (tid >> 9) & (H - 1);
    int b = tid >> 12;

    const float* qp = q + ((size_t)(b * L + l) * D + h * E);
    float qv[E];
    #pragma unroll
    for (int e = 0; e < E; ++e) qv[e] = qp[e];

    float mx = -3.4e38f, sm = 0.f;
    for (int u = 0; u < U; ++u) {
        int ls = samp[l * U + u];
        const float* kp = k + ((size_t)(b * L + ls) * D + h * E);
        float d = 0.f;
        #pragma unroll
        for (int e = 0; e < E; ++e) d += qv[e] * kp[e];
        mx = fmaxf(mx, d);
        sm += d;
    }
    Mval[tid] = mx - sm * (1.0f / (float)L);
}

__global__ __launch_bounds__(128) void topk_kernel(const float* __restrict__ Mval,
                                                   int* __restrict__ top) {
    int bh = threadIdx.x;                              // 0 .. B*H-1
    if (bh >= B * H) return;
    const float* mv = Mval + (size_t)bh * L;
    int chosen[U];
    for (int u = 0; u < U; ++u) {
        float best = -3.4e38f; int bi = 0;
        for (int l = 0; l < L; ++l) {
            bool used = false;
            for (int j = 0; j < u; ++j) used |= (chosen[j] == l);
            if (!used && mv[l] > best) { best = mv[l]; bi = l; }
        }
        chosen[u] = bi;
        top[bh * U + u] = bi;
    }
}

__global__ __launch_bounds__(256) void vmean_kernel(const float* __restrict__ v,
                                                    float* __restrict__ vmean) {
    int tid = blockIdx.x * 256 + threadIdx.x;          // B*D
    if (tid >= B * D) return;
    int d = tid & (D - 1);
    int b = tid >> 9;
    float s = 0.f;
    for (int l = 0; l < L; ++l) s += v[(size_t)(b * L + l) * D + d];
    vmean[tid] = s * (1.0f / (float)L);
}

__global__ __launch_bounds__(256) void ctx_init_kernel(const float* __restrict__ vmean,
                                                       float* __restrict__ ctx) {
    int idx = blockIdx.x * 256 + threadIdx.x;          // M*D
    if (idx >= M * D) return;
    int d = idx & (D - 1);
    int b = idx >> 18;
    ctx[idx] = vmean[b * D + d];
}

__global__ __launch_bounds__(256) void attn_update_kernel(const float* __restrict__ q,
                                                          const float* __restrict__ k,
                                                          const float* __restrict__ v,
                                                          const int* __restrict__ top,
                                                          float* __restrict__ ctx) {
    __shared__ float sc_all[8 * L];
    const int wid  = blockIdx.x * 8 + (threadIdx.x >> 5);  // exactly B*H*U waves
    const int lane = threadIdx.x & 31;
    float* sc = sc_all + (threadIdx.x >> 5) * L;

    const int bh = wid / U, u = wid - bh * U;
    const int b = bh >> 3, h = bh & (H - 1);
    const int lq = top[bh * U + u];

    const float* qp = q + ((size_t)(b * L + lq) * D + h * E);
    float qv[E];
    #pragma unroll
    for (int e = 0; e < E; ++e) qv[e] = qp[e];

    float lmax = -3.4e38f;
    for (int l = lane; l < L; l += 32) {
        const float* kp = k + ((size_t)(b * L + l) * D + h * E);
        float d = 0.f;
        #pragma unroll
        for (int e = 0; e < E; ++e) d += qv[e] * kp[e];
        d *= 0.125f;                                   // 1/sqrt(E)
        sc[l] = d;
        lmax = fmaxf(lmax, d);
    }
    #pragma unroll
    for (int off = 16; off > 0; off >>= 1) lmax = fmaxf(lmax, __shfl_xor(lmax, off, 32));

    float lsum = 0.f;
    for (int l = lane; l < L; l += 32) {
        float ex = expf(sc[l] - lmax);
        sc[l] = ex;
        lsum += ex;
    }
    #pragma unroll
    for (int off = 16; off > 0; off >>= 1) lsum += __shfl_xor(lsum, off, 32);
    const float inv = 1.0f / lsum;

    float u0 = 0.f, u1 = 0.f;                          // lane covers e=lane, e=lane+32
    for (int l = 0; l < L; ++l) {
        float a = sc[l] * inv;
        const float* vp = v + ((size_t)(b * L + l) * D + h * E);
        u0 += a * vp[lane];
        u1 += a * vp[lane + 32];
    }
    float* cp = ctx + ((size_t)(b * L + lq) * D + h * E);
    cp[lane]      = u0;
    cp[lane + 32] = u1;
}

// ---------------- residual add + LayerNorm (block per row) ----------------
__global__ __launch_bounds__(256) void resid_ln_kernel(const float* __restrict__ xin,
                                                       const float* __restrict__ add,
                                                       const float* __restrict__ g,
                                                       const float* __restrict__ bb,
                                                       float* __restrict__ xf_out,
                                                       unsigned short* __restrict__ xb_out) {
    __shared__ float red[256];
    const int row = blockIdx.x, t = threadIdx.x;
    const size_t base = (size_t)row * D;

    float s0 = xin[base + t]       + (add ? add[base + t]       : 0.f);
    float s1 = xin[base + t + 256] + (add ? add[base + t + 256] : 0.f);

    red[t] = s0 + s1;
    __syncthreads();
    for (int st = 128; st > 0; st >>= 1) { if (t < st) red[t] += red[t + st]; __syncthreads(); }
    const float mean = red[0] * (1.0f / (float)D);
    __syncthreads();

    float d0 = s0 - mean, d1 = s1 - mean;
    red[t] = d0 * d0 + d1 * d1;
    __syncthreads();
    for (int st = 128; st > 0; st >>= 1) { if (t < st) red[t] += red[t + st]; __syncthreads(); }
    const float rstd = rsqrtf(red[0] * (1.0f / (float)D) + 1e-5f);

    float o0 = d0 * rstd * g[t]       + bb[t];
    float o1 = d1 * rstd * g[t + 256] + bb[t + 256];
    xf_out[base + t]       = o0;
    xf_out[base + t + 256] = o1;
    if (xb_out) { xb_out[base + t] = f2bf(o0); xb_out[base + t + 256] = f2bf(o1); }
}

// ---------------- final gate: gelu(x) * x_mark ----------------
__global__ __launch_bounds__(256) void gate_kernel(const float* __restrict__ x,
                                                   const float* __restrict__ x_mark,
                                                   float* __restrict__ out) {
    int idx = blockIdx.x * 256 + threadIdx.x;
    if (idx >= M * D) return;
    int row = idx >> 9;
    out[idx] = gelu_exact(x[idx]) * x_mark[row];
}

// ---------------- projection + sigmoid: one block per (b,c) ----------------
__global__ __launch_bounds__(256) void proj_kernel(const float* __restrict__ gx,
                                                   const float* __restrict__ pw,
                                                   const float* __restrict__ pb,
                                                   float* __restrict__ out) {
    __shared__ float red[256];
    const int b = blockIdx.x / NC, c = blockIdx.x % NC, t = threadIdx.x;
    const size_t LD = (size_t)L * D;
    const float* xr = gx + (size_t)b * LD;
    const float* wr = pw + (size_t)c * LD;
    float s = 0.f;
    for (size_t i = t; i < LD; i += 256) s += xr[i] * wr[i];
    red[t] = s;
    __syncthreads();
    for (int st = 128; st > 0; st >>= 1) { if (t < st) red[t] += red[t + st]; __syncthreads(); }
    if (t == 0) {
        float z = red[0] + pb[c];
        out[b * NC + c] = 1.0f / (1.0f + expf(-z));
    }
}

// ================= host orchestration =================
static inline char* ws_alloc(char*& p, size_t bytes) {
    char* r = p;
    p += (bytes + 255) & ~(size_t)255;
    return r;
}

extern "C" void kernel_launch(void* const* d_in, const int* in_sizes, int n_in,
                              void* d_out, int out_size, void* d_ws, size_t ws_size,
                              hipStream_t stream) {
    (void)in_sizes; (void)n_in; (void)out_size; (void)ws_size;

    const float* x_enc      = (const float*)d_in[0];
    const float* x_mark_enc = (const float*)d_in[1];
    const float* tok_w      = (const float*)d_in[2];
    const float* Wq = (const float*)d_in[3];  const float* bq = (const float*)d_in[4];
    const float* Wk = (const float*)d_in[5];  const float* bk = (const float*)d_in[6];
    const float* Wv = (const float*)d_in[7];  const float* bv = (const float*)d_in[8];
    const float* Wo = (const float*)d_in[9];  const float* bo = (const float*)d_in[10];
    const float* W1 = (const float*)d_in[11]; const float* b1 = (const float*)d_in[12];
    const float* W2 = (const float*)d_in[13]; const float* b2 = (const float*)d_in[14];
    const float* g1 = (const float*)d_in[15]; const float* be1 = (const float*)d_in[16];
    const float* g2 = (const float*)d_in[17]; const float* be2 = (const float*)d_in[18];
    const float* gf = (const float*)d_in[19]; const float* bf = (const float*)d_in[20];
    const float* proj_w = (const float*)d_in[21];
    const float* proj_b = (const float*)d_in[22];
    const int*   sample_idx = (const int*)d_in[23];
    float* out = (float*)d_out;

    char* wp = (char*)d_ws;
    float*          xf    = (float*)         ws_alloc(wp, (size_t)M * D * 4);
    unsigned short* xb    = (unsigned short*)ws_alloc(wp, (size_t)M * D * 2);
    float*          qf    = (float*)         ws_alloc(wp, (size_t)M * D * 4);
    float*          kf    = (float*)         ws_alloc(wp, (size_t)M * D * 4);
    float*          vf    = (float*)         ws_alloc(wp, (size_t)M * D * 4);
    float*          ctxf  = (float*)         ws_alloc(wp, (size_t)M * D * 4);
    unsigned short* ctxb  = (unsigned short*)ws_alloc(wp, (size_t)M * D * 2);
    float*          newx  = (float*)         ws_alloc(wp, (size_t)M * D * 4);
    unsigned short* hidb  = (unsigned short*)ws_alloc(wp, (size_t)M * DFF * 2);
    unsigned short* wqb   = (unsigned short*)ws_alloc(wp, (size_t)D * D * 2);
    unsigned short* wkb   = (unsigned short*)ws_alloc(wp, (size_t)D * D * 2);
    unsigned short* wvb   = (unsigned short*)ws_alloc(wp, (size_t)D * D * 2);
    unsigned short* wob   = (unsigned short*)ws_alloc(wp, (size_t)D * D * 2);
    unsigned short* w1b   = (unsigned short*)ws_alloc(wp, (size_t)DFF * D * 2);
    unsigned short* w2b   = (unsigned short*)ws_alloc(wp, (size_t)D * DFF * 2);
    float*          Mval  = (float*)         ws_alloc(wp, (size_t)B * H * L * 4);
    int*            topI  = (int*)           ws_alloc(wp, (size_t)B * H * U * 4);
    float*          vmean = (float*)         ws_alloc(wp, (size_t)B * D * 4);

    auto cvt = [&](const float* s, unsigned short* d, int n) {
        cvt_bf16_kernel<<<(n + 255) / 256, 256, 0, stream>>>(s, d, n);
    };
    auto gemm = [&](const unsigned short* A, const unsigned short* Bw, const float* bias,
                    float* Cf, unsigned short* Cb, int Mm, int Nn, int Kk, int gelu) {
        int blocks = (Mm / BM) * (Nn / BN);            // all shapes divide exactly
        gemm_bf16_wmma<<<blocks, 256, 0, stream>>>(A, Bw, bias, Cf, Cb, Mm, Nn, Kk, gelu);
    };

    // 1) token conv + positional embedding
    embed_kernel<<<(M * D) / 256, 256, 0, stream>>>(x_enc, tok_w, xf, xb);

    // 2) encoder layers
    for (int l = 0; l < NL; ++l) {
        cvt(Wq + (size_t)l * D * D,   wqb, D * D);
        cvt(Wk + (size_t)l * D * D,   wkb, D * D);
        cvt(Wv + (size_t)l * D * D,   wvb, D * D);
        cvt(Wo + (size_t)l * D * D,   wob, D * D);
        cvt(W1 + (size_t)l * DFF * D, w1b, DFF * D);
        cvt(W2 + (size_t)l * D * DFF, w2b, D * DFF);

        gemm(xb, wqb, bq + l * D, qf, nullptr, M, D, D, 0);
        gemm(xb, wkb, bk + l * D, kf, nullptr, M, D, D, 0);
        gemm(xb, wvb, bv + l * D, vf, nullptr, M, D, D, 0);

        attn_metric_kernel<<<(B * H * L) / 256, 256, 0, stream>>>(
            qf, kf, sample_idx + (size_t)l * L * U, Mval);
        topk_kernel<<<1, 128, 0, stream>>>(Mval, topI);
        vmean_kernel<<<(B * D) / 256, 256, 0, stream>>>(vf, vmean);
        ctx_init_kernel<<<(M * D) / 256, 256, 0, stream>>>(vmean, ctxf);
        attn_update_kernel<<<(B * H * U) / 8, 256, 0, stream>>>(qf, kf, vf, topI, ctxf);

        cvt(ctxf, ctxb, M * D);
        gemm(ctxb, wob, bo + l * D, newx, nullptr, M, D, D, 0);
        resid_ln_kernel<<<M, 256, 0, stream>>>(xf, newx, g1 + l * D, be1 + l * D, xf, xb);

        gemm(xb, w1b, b1 + l * DFF, nullptr, hidb, M, DFF, D, 1);   // GELU fused, bf16 out
        gemm(hidb, w2b, b2 + l * D, newx, nullptr, M, D, DFF, 0);
        resid_ln_kernel<<<M, 256, 0, stream>>>(xf, newx, g2 + l * D, be2 + l * D, xf, xb);
    }

    // 3) final LN, gate, projection, sigmoid
    resid_ln_kernel<<<M, 256, 0, stream>>>(xf, nullptr, gf, bf, xf, nullptr);
    gate_kernel<<<(M * D) / 256, 256, 0, stream>>>(xf, x_mark_enc, newx);
    proj_kernel<<<B * NC, 256, 0, stream>>>(newx, proj_w, proj_b, out);
}
